// attention_59450937312095
// MI455X (gfx1250) — matrix operand order repaired and checked by reading, not compiled, not run on hardware
//
#include <hip/hip_runtime.h>
#include <hip/hip_bf16.h>
#include <math.h>

// ---------------------------------------------------------------------------
// Fused "region-masked semantic MLP" for MI455X (gfx1250, wave32, WMMA).
//
// Pipeline: repack_weights -> nms -> mask -> fused GEMM chain (WMMA f16).
// Problem sizes: B=4, C=64, H=W=128, D=256, N=100 boxes, MAX_DET=10.
// ---------------------------------------------------------------------------

typedef __attribute__((ext_vector_type(16))) _Float16 v16h;
typedef __attribute__((ext_vector_type(8)))  _Float16 v8h;
typedef __attribute__((ext_vector_type(8)))  float    v8f;

#define Bn   4
#define Cn   64
#define Hn   128
#define Wn   128
#define Dn   256
#define HWn  (Hn * Wn)            // 16384
#define ROWS (Bn * HWn)           // 65536
#define NBOX 100
#define MAXD 10
#define TOPK 5

// workspace layout (f16 fragments first, then floats, then mask bytes)
#define WSEM_F16_OFF 0            // 32 frags  * 512 f16 = 16384
#define W1_F16_OFF   16384        // 128 frags * 512 f16 = 65536
#define W2_F16_OFF   81920        // 128 frags * 512 f16 = 65536
#define WFRAG_TOTAL  147456       // f16 elements
#define KEPT_B_OFF   ((size_t)WFRAG_TOTAL * 2)            // 200 floats
#define VALID_B_OFF  (KEPT_B_OFF + 200 * 4)               // 40 floats
#define MASK_B_OFF   (VALID_B_OFF + 40 * 4 + 96)          // 65536 bytes (aligned)

// d_out layout (float elements): seg(65536) | mask(65536) | kept(200) | valid(40)
#define OUT_SEG   0
#define OUT_MASK  65536
#define OUT_KEPT  131072
#define OUT_VALID 131272

// ---------------------------------------------------------------------------
// 1) Repack W_sem/W1/W2 (f32 row-major KxN) into f16 WMMA B-fragments.
//    Fragment (kt,nf): flat = frag*512 + L*16 + e ;
//    K = kt*32 + (L/16)*8 + (e%8) + (e/8)*16, N = nf*16 + (L%16). This matches
//    the 16-bit B layout (lanes 0-15: K 0-7,16-23; lanes 16-31: K 8-15,24-31).
// ---------------------------------------------------------------------------
__global__ __launch_bounds__(256) void repack_weights_kernel(
    const float* __restrict__ Wsem, const float* __restrict__ W1,
    const float* __restrict__ W2, _Float16* __restrict__ wfrag)
{
  int gid = blockIdx.x * 256 + threadIdx.x;
  if (gid >= WFRAG_TOTAL) return;
  const float* src;
  int local;
  if (gid < 16384)      { src = Wsem; local = gid; }
  else if (gid < 81920) { src = W1;   local = gid - 16384; }
  else                  { src = W2;   local = gid - 81920; }
  int frag = local >> 9;
  int r    = local & 511;
  int L    = r >> 4;
  int e    = r & 15;
  int kt   = frag >> 4;
  int nf   = frag & 15;
  int k = kt * 32 + (L >> 4) * 8 + (e & 7) + (e >> 3) * 16;
  int n = nf * 16 + (L & 15);
  wfrag[gid] = (_Float16)src[k * Dn + n];
}

// ---------------------------------------------------------------------------
// 2) NMS: one block per batch, serial argmax (100 boxes, 10 steps — trivial).
// ---------------------------------------------------------------------------
__global__ __launch_bounds__(128) void nms_kernel(
    const float* __restrict__ boxes,   // (B, 100, 5)
    float* __restrict__ ws_kept, float* __restrict__ ws_valid,
    float* __restrict__ out_kept, float* __restrict__ out_valid)
{
  const int b = blockIdx.x, tid = threadIdx.x;
  __shared__ float bx[NBOX][5];
  __shared__ float area[NBOX];
  __shared__ int   cand[NBOX];
  __shared__ int   idxs[MAXD];
  __shared__ int   hasv[MAXD];
  __shared__ int   useThres;
  if (tid == 0) useThres = 0;
  __syncthreads();
  if (tid < NBOX) {
    const float* p = boxes + (b * NBOX + tid) * 5;
    for (int j = 0; j < 5; ++j) bx[tid][j] = p[j];
    float w = fmaxf(bx[tid][2] - bx[tid][0], 0.0f);
    float h = fmaxf(bx[tid][3] - bx[tid][1], 0.0f);
    area[tid] = w * h;
    if (bx[tid][4] > 0.2f) atomicOr(&useThres, 1);
  }
  __syncthreads();
  if (tid < NBOX) cand[tid] = useThres ? (bx[tid][4] > 0.2f ? 1 : 0) : 1;
  __syncthreads();
  for (int t = 0; t < MAXD; ++t) {
    if (tid == 0) {
      float best = -INFINITY; int bi = 0;
      for (int j = 0; j < NBOX; ++j)
        if (cand[j] && bx[j][4] > best) { best = bx[j][4]; bi = j; }
      idxs[t] = bi;
      hasv[t] = (best > -INFINITY) ? 1 : 0;
    }
    __syncthreads();
    if (hasv[t] && tid < NBOX) {
      int i = idxs[t];
      float lx = fmaxf(bx[i][0], bx[tid][0]);
      float ly = fmaxf(bx[i][1], bx[tid][1]);
      float rx = fminf(bx[i][2], bx[tid][2]);
      float ry = fminf(bx[i][3], bx[tid][3]);
      float inter = fmaxf(rx - lx, 0.0f) * fmaxf(ry - ly, 0.0f);
      float iou = inter / (area[i] + area[tid] - inter + 1e-9f);
      if (iou > 0.4f || tid == i) cand[tid] = 0;
    }
    __syncthreads();
  }
  if (tid < MAXD) {
    int i = idxs[tid];
    float k0 = bx[i][0], k1 = bx[i][1], k2 = bx[i][2], k3 = bx[i][3], k4 = bx[i][4];
    int v = hasv[tid] && (k2 - k0 >= 1.0f) && (k3 - k1 >= 1.0f) &&
            (useThres || tid < TOPK);
    float* kd = ws_kept + (b * MAXD + tid) * 5;
    kd[0] = k0; kd[1] = k1; kd[2] = k2; kd[3] = k3; kd[4] = k4;
    ws_valid[b * MAXD + tid] = v ? 1.0f : 0.0f;
    float* ko = out_kept + (b * MAXD + tid) * 5;
    ko[0] = k0; ko[1] = k1; ko[2] = k2; ko[3] = k3; ko[4] = k4;
    out_valid[b * MAXD + tid] = v ? 1.0f : 0.0f;
  }
}

// ---------------------------------------------------------------------------
// 3) Per-pixel inside mask: mask_maps output + byte mask for the fused GEMM.
// ---------------------------------------------------------------------------
__global__ __launch_bounds__(256) void mask_kernel(
    const float* __restrict__ ws_kept, const float* __restrict__ ws_valid,
    float* __restrict__ out_mask, unsigned char* __restrict__ ws_mask)
{
  int g = blockIdx.x * 256 + threadIdx.x;          // 0..65535
  int b = g >> 14, p = g & (HWn - 1);
  int h = p >> 7,  w = p & (Wn - 1);
  __shared__ int X1[MAXD], Y1[MAXD], X2[MAXD], Y2[MAXD], V[MAXD];
  if (threadIdx.x < MAXD) {
    int t = threadIdx.x;
    const float* k = ws_kept + (b * MAXD + t) * 5;
    int x1 = (int)floorf(k[0] + 0.5f);
    int y1 = (int)floorf(k[1] + 0.5f);
    int x2 = (int)floorf(k[2] + 0.5f);
    int y2 = (int)floorf(k[3] + 0.5f);
    X1[t] = min(max(x1, 0), Wn); Y1[t] = min(max(y1, 0), Hn);
    X2[t] = min(max(x2, 0), Wn); Y2[t] = min(max(y2, 0), Hn);
    V[t]  = ws_valid[b * MAXD + t] > 0.5f;
  }
  __syncthreads();
  int ins = 0;
  for (int t = 0; t < MAXD; ++t)
    ins |= (V[t] && w >= X1[t] && w < X2[t] && h >= Y1[t] && h < Y2[t]);
  out_mask[g] = ins ? 0.0f : 1.0f;   // mask_maps = ~inside_any
  ws_mask[g]  = (unsigned char)ins;
}

// ---------------------------------------------------------------------------
// 4) Fused: seman GEMM -> mask+bias -> MLP (relu GEMM x2) -> 256-dot -> seg.
//    64 rows/block, 8 waves: wave = (mtile 0..3) x (nhalf 0..1).
// ---------------------------------------------------------------------------
#define SMA_S 72    // 64 + 8 f16 pad (conflict-free ds_load_b128)
#define ACT_S 264   // 256 + 8 f16 pad

__global__ __launch_bounds__(256) void fused_mlp_kernel(
    const float* __restrict__ x,
    const _Float16* __restrict__ wfrag,
    const float* __restrict__ b_sem, const float* __restrict__ b1,
    const float* __restrict__ b2, const float* __restrict__ W3,
    const float* __restrict__ b3,
    const unsigned char* __restrict__ ws_mask,
    float* __restrict__ out_seg)
{
  __shared__ _Float16 smA[64 * SMA_S];
  __shared__ _Float16 smAct[64 * ACT_S];
  __shared__ float smW3[Dn];
  __shared__ unsigned char smMask[64];

  const int tid  = threadIdx.x;
  const int lane = tid & 31;
  const int wv   = tid >> 5;
  const int mtile = wv & 3;           // 16-row tile within the 64-row block
  const int nhalf = wv >> 2;          // N 0..127 or 128..255
  const int nbase = nhalf * 128;
  const int row0 = blockIdx.x * 64;
  const int b  = row0 >> 14;
  const int p0 = row0 & (HWn - 1);

  // L2 warm-up of W1 fragments while we stage x (emits global_prefetch_b8)
  __builtin_prefetch(wfrag + W1_F16_OFF + (size_t)tid * 256, 0, 3);

  // Stage x tile (64 pixels x 64 channels) to LDS as f16 [m][c]
  {
    const int m  = tid & 63;
    const int cb = tid >> 6;                       // 0..3
    const float* xb = x + ((size_t)b * Cn) * HWn + p0 + m;
    #pragma unroll
    for (int i = 0; i < 16; ++i) {
      int c = cb + i * 4;
      smA[m * SMA_S + c] = (_Float16)xb[(size_t)c * HWn];
    }
  }
  if (tid < 64) smMask[tid] = ws_mask[row0 + tid];
  smW3[tid] = W3[tid];
  __syncthreads();

  const int mrow = mtile * 16 + (lane & 15);
  const int half = lane >> 4;
  const int ncol = lane & 15;

  v8f acc[8];
  v8f zeroAcc;
  #pragma unroll
  for (int i = 0; i < 8; ++i) zeroAcc[i] = 0.0f;

  // ---- layer 0: seman = x_tile @ W_sem (K=64) ----
  #pragma unroll
  for (int f = 0; f < 8; ++f) acc[f] = zeroAcc;
  {
    const v16h* wb = (const v16h*)(wfrag + WSEM_F16_OFF);
    #pragma unroll
    for (int kt = 0; kt < 2; ++kt) {
      const _Float16* ap = &smA[mrow * SMA_S + kt * 32 + half * 8];
      v8h lo = *(const v8h*)ap;
      v8h hi = *(const v8h*)(ap + 16);
      v16h a = __builtin_shufflevector(lo, hi, 0,1,2,3,4,5,6,7,8,9,10,11,12,13,14,15);
      #pragma unroll
      for (int f = 0; f < 8; ++f) {
        v16h bm = wb[((kt * 16) + (nbase >> 4) + f) * 32 + lane];
        acc[f] = __builtin_amdgcn_wmma_f32_16x16x32_f16(
            false, a, false, bm, (short)0, acc[f], false, false);
      }
    }
  }
  // words = inside ? seman + bias : 0 ; store f16 to activation buffer
  #pragma unroll
  for (int f = 0; f < 8; ++f) {
    int n = nbase + f * 16 + ncol;
    float bias = b_sem[n];
    #pragma unroll
    for (int r = 0; r < 8; ++r) {
      int m = mtile * 16 + half * 8 + r;
      float v = smMask[m] ? (acc[f][r] + bias) : 0.0f;
      smAct[m * ACT_S + n] = (_Float16)v;
    }
  }
  __syncthreads();

  // ---- layers 1 & 2: h = relu(act @ W + b) in-place (barrier-separated) ----
  auto mlp_layer = [&](const v16h* wb, const float* bias_v) {
    #pragma unroll
    for (int f = 0; f < 8; ++f) acc[f] = zeroAcc;
    #pragma unroll
    for (int kt = 0; kt < 8; ++kt) {
      const _Float16* ap = &smAct[mrow * ACT_S + kt * 32 + half * 8];
      v8h lo = *(const v8h*)ap;
      v8h hi = *(const v8h*)(ap + 16);
      v16h a = __builtin_shufflevector(lo, hi, 0,1,2,3,4,5,6,7,8,9,10,11,12,13,14,15);
      #pragma unroll
      for (int f = 0; f < 8; ++f) {
        v16h bm = wb[((kt * 16) + (nbase >> 4) + f) * 32 + lane];
        acc[f] = __builtin_amdgcn_wmma_f32_16x16x32_f16(
            false, a, false, bm, (short)0, acc[f], false, false);
      }
    }
    __syncthreads();   // all waves done reading before overwrite
    #pragma unroll
    for (int f = 0; f < 8; ++f) {
      int n = nbase + f * 16 + ncol;
      float bb = bias_v[n];
      #pragma unroll
      for (int r = 0; r < 8; ++r) {
        int m = mtile * 16 + half * 8 + r;
        smAct[m * ACT_S + n] = (_Float16)fmaxf(acc[f][r] + bb, 0.0f);
      }
    }
    __syncthreads();
  };
  mlp_layer((const v16h*)(wfrag + W1_F16_OFF), b1);
  mlp_layer((const v16h*)(wfrag + W2_F16_OFF), b2);

  // ---- layer 3: seg = h2 @ W3 + b3[0] ; 4 lanes per row, quad reduction ----
  {
    const int m = tid >> 2;        // 0..63
    const int q = tid & 3;         // k quarter
    const _Float16* rowp = &smAct[m * ACT_S + q * 64];
    float s = 0.0f;
    #pragma unroll 8
    for (int k = 0; k < 64; ++k) s += (float)rowp[k] * smW3[q * 64 + k];
    s += __shfl_xor(s, 1, 32);
    s += __shfl_xor(s, 2, 32);
    if (q == 0)
      out_seg[(size_t)(p0 + m) * Bn + b] = s + b3[0];  // (H*W, B) transpose
  }
}

// ---------------------------------------------------------------------------
extern "C" void kernel_launch(void* const* d_in, const int* in_sizes, int n_in,
                              void* d_out, int out_size, void* d_ws, size_t ws_size,
                              hipStream_t stream) {
  const float* x      = (const float*)d_in[0];
  const float* boxes  = (const float*)d_in[1];
  const float* W_sem  = (const float*)d_in[2];
  const float* b_sem  = (const float*)d_in[3];
  const float* W1     = (const float*)d_in[4];
  const float* b1     = (const float*)d_in[5];
  const float* W2     = (const float*)d_in[6];
  const float* b2     = (const float*)d_in[7];
  const float* W3     = (const float*)d_in[8];
  const float* b3     = (const float*)d_in[9];

  float* out = (float*)d_out;
  char*  ws  = (char*)d_ws;

  _Float16*      wfrag    = (_Float16*)ws;
  float*         ws_kept  = (float*)(ws + KEPT_B_OFF);
  float*         ws_valid = (float*)(ws + VALID_B_OFF);
  unsigned char* ws_mask  = (unsigned char*)(ws + MASK_B_OFF);

  // 1) repack weights into WMMA fragment order (f16)
  repack_weights_kernel<<<(WFRAG_TOTAL + 255) / 256, 256, 0, stream>>>(
      W_sem, W1, W2, wfrag);

  // 2) NMS -> kept/valid (ws + out)
  nms_kernel<<<Bn, 128, 0, stream>>>(boxes, ws_kept, ws_valid,
                                     out + OUT_KEPT, out + OUT_VALID);

  // 3) inside mask -> mask_maps (out) + byte mask (ws)
  mask_kernel<<<ROWS / 256, 256, 0, stream>>>(ws_kept, ws_valid,
                                              out + OUT_MASK, ws_mask);

  // 4) fused seman -> mask -> MLP -> seg_output
  fused_mlp_kernel<<<ROWS / 64, 256, 0, stream>>>(
      x, wfrag, b_sem, b1, b2, W3, b3, ws_mask, out + OUT_SEG);
}
